// Rope_MQA_10101763080226
// MI455X (gfx1250) — matrix operand order repaired and checked by reading, not compiled, not run on hardware
//
#include <hip/hip_runtime.h>
#include <hip/hip_bf16.h>
#include <cstdint>
#include <cstddef>

typedef _Float16 half_t;
typedef __attribute__((ext_vector_type(16))) _Float16 v16h;
typedef __attribute__((ext_vector_type(8)))  float    v8f;

#define Bsz  2
#define Sseq 2048
#define Dmod 2048
#define Hn   16
#define DHd  128
#define BSr  (Bsz*Sseq)   // 4096 rows

#if defined(__has_builtin)
#  if __has_builtin(__builtin_amdgcn_tensor_load_to_lds) && __has_builtin(__builtin_amdgcn_s_wait_tensorcnt)
#    define HAVE_TDM 1
#  else
#    define HAVE_TDM 0
#  endif
#else
#  define HAVE_TDM 0
#endif

union Frag16 { v16h h; uint4 u[2]; };

static __device__ __forceinline__ v8f wmma_f16(const v16h& a, const v16h& b, const v8f& c) {
    return __builtin_amdgcn_wmma_f32_16x16x32_f16(false, a, false, b, (short)0, c, false, false);
}

#if HAVE_TDM
// 2-D TDM load: tile_w contiguous f16 elems per row, tile_h rows, row stride
// `stride_elems`; LDS destination padded by 4 DWORDs (8 halves) after every
// (1<<pad_interval_code) 8-byte units -> LDS pitch = tile_w + 8 halves.
static __device__ __forceinline__ void tdm_load_2d(half_t* lds_dst, const half_t* gsrc,
                                                   unsigned tile_w, unsigned tile_h,
                                                   unsigned stride_elems,
                                                   unsigned pad_interval_code) {
    typedef __attribute__((ext_vector_type(4))) unsigned int v4u_t;
    typedef __attribute__((ext_vector_type(8))) int v8i_t;
    typedef __attribute__((ext_vector_type(4))) int v4i_t;
    const unsigned long long ga = (unsigned long long)gsrc;
    const unsigned lds_off = (unsigned)(unsigned long long)lds_dst;  // low 32b = LDS offset
    v4u_t g0;
    g0[0] = 1u;                                             // count = 1 (valid D#)
    g0[1] = lds_off;                                        // lds_addr
    g0[2] = (unsigned)ga;                                   // global_addr[31:0]
    g0[3] = (unsigned)((ga >> 32) & 0x01ffffffULL) | 0x80000000u; // addr[56:32] | type=2
    v8i_t g1;
    g1[0] = (int)((1u << 16)                                // data_size = 2 bytes
                | (1u << 20)                                // pad_enable
                | (pad_interval_code << 22)                 // pad interval
                | (3u << 25));                              // pad_amount = 4 DWORDs
    g1[1] = (int)((tile_w & 0xffffu) << 16);                // tensor_dim0[15:0]
    g1[2] = (int)(((tile_w >> 16) & 0xffffu) | ((tile_h & 0xffffu) << 16)); // dim0 hi | dim1 lo
    g1[3] = (int)(((tile_h >> 16) & 0xffffu) | ((tile_w & 0xffffu) << 16)); // dim1 hi | tile_dim0
    g1[4] = (int)(tile_h & 0xffffu);                        // tile_dim1 (tile_dim2 = 0)
    g1[5] = (int)stride_elems;                              // tensor_dim0_stride[31:0]
    g1[6] = 0;
    g1[7] = 0;
    const v4i_t z4 = {0, 0, 0, 0};
    const v8i_t z8 = {0, 0, 0, 0, 0, 0, 0, 0};
    // clang-23 / therock arity: (g0 v4u, g1 v8i, g2 v4i, g3 v4i, g4 v8i, cpol)
    __builtin_amdgcn_tensor_load_to_lds(g0, g1, z4, z4, z8, 0);
}
#endif

// ---------------- fp32 -> f16 convert ----------------
__global__ void k_cvt(half_t* __restrict__ dst, const float* __restrict__ src, int n) {
    int i = blockIdx.x * blockDim.x + threadIdx.x;
    int stride = gridDim.x * blockDim.x;
    for (; i < n; i += stride) dst[i] = (half_t)src[i];
}

// ------------- wkv [D][2D] f32 -> wkvT [2D][D] f16 (LDS tiled) -------------
__global__ void k_wkv_transpose(half_t* __restrict__ dstT, const float* __restrict__ src) {
    __shared__ float tile[32][33];
    const int k0 = blockIdx.x * 32, n0 = blockIdx.y * 32;
    const int tx = threadIdx.x, ty = threadIdx.y;
    for (int r = 0; r < 32; r += 8)
        tile[ty + r][tx] = src[(size_t)(k0 + ty + r) * (2 * Dmod) + n0 + tx];
    __syncthreads();
    for (int r = 0; r < 32; r += 8)
        dstT[(size_t)(n0 + ty + r) * Dmod + k0 + tx] = (half_t)tile[tx][ty + r];
}

// ---------------- WMMA GEMM: C[M,N] = A[M,K] * Bt[N,K]^T ----------------
// 128x128x32 block tile, 256 threads (8 waves), wave tile 32x64 (2x4 WMMA accs),
// TDM double-buffered LDS staging (DMA of slab k+1 overlaps WMMA on slab k).
template<bool F32OUT>
__global__ __launch_bounds__(256) void k_gemm(const half_t* __restrict__ A,
                                              const half_t* __restrict__ Bt,
                                              void* __restrict__ Cout,
                                              int M, int N, int K) {
    constexpr int LDT = 40;                       // 80B pitch -> conflict-free frag reads
    alignas(16) __shared__ half_t sA[2][128 * LDT];
    alignas(16) __shared__ half_t sB[2][128 * LDT];

    const int tid  = threadIdx.x;
    const int lane = tid & 31, wid = tid >> 5;
    const int lh = lane & 15, sel = lane >> 4;
    const int wr = wid >> 1, wc = wid & 1;        // 4x2 wave grid
    const int row0 = blockIdx.y * 128;
    const int col0 = blockIdx.x * 128;
    const half_t* Ab = A + (size_t)row0 * K;
    const half_t* Bb = Bt + (size_t)col0 * K;

    auto stage = [&](int buf, int kt) {
#if HAVE_TDM
        if (wid == 0) {   // one wave issues the DMA for the whole WG
            tdm_load_2d(sA[buf], Ab + kt, 32, 128, (unsigned)K, 3);
            tdm_load_2d(sB[buf], Bb + kt, 32, 128, (unsigned)K, 3);
        }
#else
        #pragma unroll
        for (int t = 0; t < 2; ++t) {
            const int q = tid + t * 256;
            const int r = q >> 2, seg = q & 3;
            *reinterpret_cast<uint4*>(&sA[buf][r * LDT + seg * 8]) =
                *reinterpret_cast<const uint4*>(Ab + (size_t)r * K + kt + seg * 8);
            *reinterpret_cast<uint4*>(&sB[buf][r * LDT + seg * 8]) =
                *reinterpret_cast<const uint4*>(Bb + (size_t)r * K + kt + seg * 8);
        }
#endif
    };

    const v8f vzero = {0,0,0,0,0,0,0,0};
    v8f acc[2][4];
    for (int mt = 0; mt < 2; ++mt)
        for (int nt = 0; nt < 4; ++nt) acc[mt][nt] = vzero;

    stage(0, 0);
    const int nk = K / 32;
    for (int ki = 0; ki < nk; ++ki) {
        const int buf = ki & 1;
        if (ki + 1 < nk) stage(buf ^ 1, (ki + 1) * 32);
#if HAVE_TDM
        if (wid == 0) {
            if (ki + 1 < nk) __builtin_amdgcn_s_wait_tensorcnt(2); // slab ki complete
            else             __builtin_amdgcn_s_wait_tensorcnt(0);
        }
#endif
        __syncthreads();   // publish slab ki to all waves

        Frag16 a[2], bfr[4];
        #pragma unroll
        for (int mt = 0; mt < 2; ++mt) {
            const half_t* base = &sA[buf][(wr * 32 + mt * 16 + lh) * LDT + sel * 8];
            a[mt].u[0] = *reinterpret_cast<const uint4*>(base);
            a[mt].u[1] = *reinterpret_cast<const uint4*>(base + 16);
        }
        #pragma unroll
        for (int nt = 0; nt < 4; ++nt) {
            const half_t* base = &sB[buf][(wc * 64 + nt * 16 + lh) * LDT + sel * 8];
            bfr[nt].u[0] = *reinterpret_cast<const uint4*>(base);
            bfr[nt].u[1] = *reinterpret_cast<const uint4*>(base + 16);
        }
        #pragma unroll
        for (int mt = 0; mt < 2; ++mt)
            #pragma unroll
            for (int nt = 0; nt < 4; ++nt)
                acc[mt][nt] = wmma_f16(a[mt].h, bfr[nt].h, acc[mt][nt]);
        __syncthreads();   // slab consumed; safe to overwrite next iteration
    }

    #pragma unroll
    for (int mt = 0; mt < 2; ++mt)
        #pragma unroll
        for (int nt = 0; nt < 4; ++nt)
            #pragma unroll
            for (int i = 0; i < 8; ++i) {
                const int r = row0 + wr * 32 + mt * 16 + sel * 8 + i;
                const int c = col0 + wc * 64 + nt * 16 + lh;
                if constexpr (F32OUT)
                    reinterpret_cast<float*>(Cout)[(size_t)r * N + c] = acc[mt][nt][i];
                else
                    reinterpret_cast<half_t*>(Cout)[(size_t)r * N + c] = (half_t)acc[mt][nt][i];
            }
}

// ---------------- RoPE + rearrange to [bh][s][dh] ----------------
__global__ void k_rope(const half_t* __restrict__ Qf, const half_t* __restrict__ KVf,
                       half_t* __restrict__ qr, half_t* __restrict__ kr) {
    const int tid = blockIdx.x * blockDim.x + threadIdx.x;   // B*S*H*64 threads
    const int i = tid & 63;
    const int h = (tid >> 6) & (Hn - 1);
    const int s = (tid >> 10) & (Sseq - 1);
    const int b = tid >> 21;
    // inv_freq = theta^{-i/64} ; log(10000) = 9.2103403719...
    const float ang = (float)s * __expf(-(float)i * (9.210340371976184f / 64.0f));
    float sn, c;
    sincosf(ang, &sn, &c);
    const size_t row = (size_t)b * Sseq + s;
    const half_t* qp = Qf + row * Dmod + h * DHd;
    const half_t* kp = KVf + row * (2 * Dmod) + h * DHd;
    const float q0 = (float)qp[i], q1 = (float)qp[i + 64];
    const float k0 = (float)kp[i], k1 = (float)kp[i + 64];
    const size_t ro = (((size_t)(b * Hn + h)) * Sseq + s) * DHd;
    qr[ro + i]      = (half_t)(q0 * c - q1 * sn);
    qr[ro + i + 64] = (half_t)(q1 * c + q0 * sn);
    kr[ro + i]      = (half_t)(k0 * c - k1 * sn);
    kr[ro + i + 64] = (half_t)(k1 * c + k0 * sn);
}

// ---------------- V -> vrT [bh][dh][S] f16 (LDS tiled transpose) ----------------
__global__ void k_vtrans(const half_t* __restrict__ KVf, half_t* __restrict__ vrT) {
    __shared__ half_t tile[32][40];
    const int bh = blockIdx.z, b = bh >> 4, h = bh & 15;
    const int s0 = blockIdx.y * 32, d0 = blockIdx.x * 32;
    const int tx = threadIdx.x, ty = threadIdx.y;
    for (int r = 0; r < 32; r += 8) {
        const int s = s0 + ty + r;
        tile[ty + r][tx] = KVf[((size_t)(b * Sseq + s)) * (2 * Dmod) + Dmod + h * DHd + d0 + tx];
    }
    __syncthreads();
    for (int r = 0; r < 32; r += 8) {
        const int d = d0 + ty + r;
        vrT[((size_t)bh * DHd + d) * Sseq + s0 + tx] = tile[tx][ty + r];
    }
}

// ---------------- causal flash attention ----------------
// Per-wave 16 q rows, KV blocks of 32, waves in lockstep; K/V blocks staged
// once per WG into LDS by the TDM (double-buffered), shared by all 8 waves.
__global__ __launch_bounds__(256) void k_attn(const half_t* __restrict__ qr,
                                              const half_t* __restrict__ kr,
                                              const half_t* __restrict__ vrT,
                                              half_t* __restrict__ ao) {
    constexpr int KP = 136;   // K tile LDS pitch (128 + 8 pad halves)
    constexpr int VP = 40;    // V tile LDS pitch (32 + 8 pad halves)
    alignas(16) __shared__ half_t sK[2][32 * KP];
    alignas(16) __shared__ half_t sV[2][128 * VP];
    __shared__ float  sS[8][16 * 32];
    alignas(16) __shared__ half_t sP[8][16 * 40];
    __shared__ float  sAl[8][16];
    __shared__ float  sL[8][16];

    const int tid = threadIdx.x;
    const int lane = tid & 31, w = tid >> 5;
    const int lh = lane & 15, sel = lane >> 4;
    const int qbase = blockIdx.x * 128;
    const int qrow0 = qbase + w * 16;
    const int bh = blockIdx.y;
    const int b = bh >> 4, h = bh & 15;
    const float scale = 0.08838834764831845f;   // 1/sqrt(128)

    const half_t* kbase = kr + (size_t)bh * Sseq * DHd;    // [S][dh]
    const half_t* vbase = vrT + (size_t)bh * DHd * Sseq;   // [dh][S]

    auto stage = [&](int buf, int kv0) {
#if HAVE_TDM
        if (w == 0) {
            tdm_load_2d(sK[buf], kbase + (size_t)kv0 * DHd, 128, 32, DHd, 5);
            tdm_load_2d(sV[buf], vbase + kv0,               32, 128, Sseq, 3);
        }
#else
        #pragma unroll
        for (int t = 0; t < 2; ++t) {
            const int q = tid + t * 256;
            { const int r = q >> 4, seg = q & 15;   // K: 32 rows x 16 chunks
              *reinterpret_cast<uint4*>(&sK[buf][r * KP + seg * 8]) =
                  *reinterpret_cast<const uint4*>(kbase + (size_t)(kv0 + r) * DHd + seg * 8); }
            { const int r = q >> 2, seg = q & 3;    // V: 128 rows x 4 chunks
              *reinterpret_cast<uint4*>(&sV[buf][r * VP + seg * 8]) =
                  *reinterpret_cast<const uint4*>(vbase + (size_t)r * Sseq + kv0 + seg * 8); }
        }
#endif
    };

    // load Q fragments once (A-matrix layout, 4 k-steps of 32 over dh=128)
    const half_t* qp = qr + ((size_t)bh * Sseq + qrow0) * DHd;
    Frag16 aq[4];
    #pragma unroll
    for (int kb = 0; kb < 4; ++kb) {
        const half_t* base = qp + (size_t)lh * DHd + kb * 32 + sel * 8;
        aq[kb].u[0] = *reinterpret_cast<const uint4*>(base);
        aq[kb].u[1] = *reinterpret_cast<const uint4*>(base + 16);
    }

    const v8f vzero = {0,0,0,0,0,0,0,0};
    v8f o[8];
    #pragma unroll
    for (int nt = 0; nt < 8; ++nt) o[nt] = vzero;
    float mreg = -1e30f, lreg = 0.0f;

    const int nblk = qbase / 32 + 4;   // causal: kv <= qbase+127
    stage(0, 0);
    for (int j = 0; j < nblk; ++j) {
        const int kv0 = j * 32;
        const int buf = j & 1;
        if (j + 1 < nblk) stage(buf ^ 1, kv0 + 32);
#if HAVE_TDM
        if (w == 0) {
            if (j + 1 < nblk) __builtin_amdgcn_s_wait_tensorcnt(2);  // block j complete
            else              __builtin_amdgcn_s_wait_tensorcnt(0);
        }
#endif
        __syncthreads();   // publish block j

        const bool act = (kv0 <= qrow0 + 15);
        const half_t* kc = sK[buf];
        const half_t* vc = sV[buf];
        if (act) {
            v8f sc[2]; sc[0] = vzero; sc[1] = vzero;
            #pragma unroll
            for (int nt = 0; nt < 2; ++nt) {
                Frag16 bk[4];                     // distinct regs -> batched DS reads
                #pragma unroll
                for (int kb = 0; kb < 4; ++kb) {
                    const half_t* base = kc + (nt * 16 + lh) * KP + kb * 32 + sel * 8;
                    bk[kb].u[0] = *reinterpret_cast<const uint4*>(base);
                    bk[kb].u[1] = *reinterpret_cast<const uint4*>(base + 16);
                }
                #pragma unroll
                for (int kb = 0; kb < 4; ++kb)
                    sc[nt] = wmma_f16(aq[kb].h, bk[kb].h, sc[nt]);
            }
            // scaled + causal-masked scores into per-wave LDS (C layout -> row-major)
            #pragma unroll
            for (int nt = 0; nt < 2; ++nt)
                #pragma unroll
                for (int i = 0; i < 8; ++i) {
                    const int m = sel * 8 + i, c = nt * 16 + lh;
                    float v = sc[nt][i] * scale;
                    if (kv0 + c > qrow0 + m) v = -1e30f;
                    sS[w][m * 32 + c] = v;
                }
        }
        __syncthreads();
        if (act && lane < 16) {   // online softmax, row = lane
            const int r = lane;
            float mx = mreg;
            for (int c = 0; c < 32; ++c) mx = fmaxf(mx, sS[w][r * 32 + c]);
            const float alpha = __expf(mreg - mx);
            float sum = 0.0f;
            for (int c = 0; c < 32; ++c) {
                const float p = __expf(sS[w][r * 32 + c] - mx);
                sum += p;
                sP[w][r * 40 + c] = (half_t)p;
            }
            lreg = lreg * alpha + sum;
            mreg = mx;
            sAl[w][r] = alpha;
        }
        __syncthreads();
        if (act) {
            // rescale accumulators, then O += P * V
            #pragma unroll
            for (int i = 0; i < 8; ++i) {
                const float a = sAl[w][sel * 8 + i];
                #pragma unroll
                for (int nt = 0; nt < 8; ++nt) o[nt][i] *= a;
            }
            Frag16 pf;
            const half_t* pb = &sP[w][lh * 40];
            pf.u[0] = *reinterpret_cast<const uint4*>(pb + sel * 8);
            pf.u[1] = *reinterpret_cast<const uint4*>(pb + sel * 8 + 16);
            #pragma unroll
            for (int half8 = 0; half8 < 2; ++half8) {
                Frag16 bv[4];
                #pragma unroll
                for (int q4 = 0; q4 < 4; ++q4) {
                    const int nt = half8 * 4 + q4;
                    const half_t* base = vc + (nt * 16 + lh) * VP + sel * 8;
                    bv[q4].u[0] = *reinterpret_cast<const uint4*>(base);
                    bv[q4].u[1] = *reinterpret_cast<const uint4*>(base + 16);
                }
                #pragma unroll
                for (int q4 = 0; q4 < 4; ++q4)
                    o[half8 * 4 + q4] = wmma_f16(pf.h, bv[q4].h, o[half8 * 4 + q4]);
            }
        }
        __syncthreads();   // block j consumed; buffers/sS/sP safe to rewrite
    }

    if (lane < 16) sL[w][lane] = lreg;
    __syncthreads();
    // write back to ao [B*S][D] (undoes head split)
    #pragma unroll
    for (int i = 0; i < 8; ++i) {
        const int m = sel * 8 + i;
        const float inv = 1.0f / sL[w][m];
        const size_t row = (size_t)(b * Sseq + qrow0 + m);
        #pragma unroll
        for (int nt = 0; nt < 8; ++nt) {
            const int d = h * DHd + nt * 16 + lh;
            ao[row * Dmod + d] = (half_t)(o[nt][i] * inv);
        }
    }
}

extern "C" void kernel_launch(void* const* d_in, const int* in_sizes, int n_in,
                              void* d_out, int out_size, void* d_ws, size_t ws_size,
                              hipStream_t stream) {
    (void)in_sizes; (void)n_in; (void)out_size; (void)ws_size;
    const float* x   = (const float*)d_in[0];   // [B,S,D]
    const float* wq  = (const float*)d_in[1];   // [D,D]
    const float* wkv = (const float*)d_in[2];   // [D,2D]
    const float* wo  = (const float*)d_in[3];   // [D,D]
    float* out = (float*)d_out;                 // [B,S,D] fp32

    char* ws = (char*)d_ws;
    const size_t MB = 1024 * 1024;
    half_t* xh   = (half_t*)(ws +   0 * MB);   // 16 MB  x f16 [BS][D]
    half_t* wqh  = (half_t*)(ws +  16 * MB);   //  8 MB
    half_t* wkvT = (half_t*)(ws +  24 * MB);   // 16 MB  [2D][D]
    half_t* woh  = (half_t*)(ws +  40 * MB);   //  8 MB
    half_t* Qf   = (half_t*)(ws +  48 * MB);   // 16 MB  [BS][D]
    half_t* KVf  = (half_t*)(ws +  64 * MB);   // 32 MB  [BS][2D]
    half_t* qrb  = (half_t*)(ws +  96 * MB);   // 16 MB  [bh][S][dh]
    half_t* krb  = (half_t*)(ws + 112 * MB);   // 16 MB
    half_t* vrT  = xh;                          // xh dead after GEMM2
    half_t* ao   = Qf;                          // Qf dead after RoPE

    k_cvt<<<2048, 256, 0, stream>>>(xh, x, BSr * Dmod);
    k_cvt<<<1024, 256, 0, stream>>>(wqh, wq, Dmod * Dmod);
    k_cvt<<<1024, 256, 0, stream>>>(woh, wo, Dmod * Dmod);
    k_wkv_transpose<<<dim3(Dmod / 32, 2 * Dmod / 32), dim3(32, 8), 0, stream>>>(wkvT, wkv);

    k_gemm<false><<<dim3(Dmod / 128, BSr / 128), 256, 0, stream>>>(xh, wqh, (void*)Qf, BSr, Dmod, Dmod);
    k_gemm<false><<<dim3(2 * Dmod / 128, BSr / 128), 256, 0, stream>>>(xh, wkvT, (void*)KVf, BSr, 2 * Dmod, Dmod);

    k_rope<<<(Bsz * Sseq * Hn * 64) / 256, 256, 0, stream>>>(Qf, KVf, qrb, krb);
    k_vtrans<<<dim3(DHd / 32, Sseq / 32, Bsz * Hn), dim3(32, 8), 0, stream>>>(KVf, vrT);

    k_attn<<<dim3(Sseq / 128, Bsz * Hn), 256, 0, stream>>>(qrb, krb, vrT, ao);

    k_gemm<true><<<dim3(Dmod / 128, BSr / 128), 256, 0, stream>>>(ao, woh, (void*)out, BSr, Dmod, Dmod);
}
